// ALIGNN_7275674599776
// MI455X (gfx1250) — compile-verified, matrix operand order checked by reading, not run on hardware
//
#include <hip/hip_runtime.h>
#include <cstdint>
#include <cstddef>

// ---------------- model constants ----------------
#define NNODE 2048
#define NEDGE 131072
#define NLINE 524288
#define DM    128
#define NH    4
#define DH    32
#define NLAY  4
#define FPI   3.14159265358979323846f

// ---------------- types ----------------
typedef __bf16 bf16t;
typedef bf16t v16bf __attribute__((ext_vector_type(16)));
typedef float v8f  __attribute__((ext_vector_type(8)));
typedef unsigned short u16;

__device__ __forceinline__ u16 f2bf(float f) {
  unsigned int u = __float_as_uint(f);
  unsigned int r = 0x7FFFu + ((u >> 16) & 1u);
  return (u16)((u + r) >> 16);
}
__device__ __forceinline__ float siluf(float x) { return x / (1.0f + __expf(-x)); }
__device__ __forceinline__ v16bf ldv16(const u16* p) {
  return *reinterpret_cast<const v16bf*>(p);
}
__device__ __forceinline__ v8f vzero() {
  v8f z;
#pragma unroll
  for (int i = 0; i < 8; ++i) z[i] = 0.0f;
  return z;
}

// ---------------- small elementwise kernels ----------------
// f32 -> bf16, transposed: src is [layers][Ksrc][N] row-major, dst is [layers][N][Kdst]
__global__ void k_f2bt(u16* dst, const float* __restrict__ src, int Ksrc, int N, int Kdst, int layers) {
  int t = blockIdx.x * blockDim.x + threadIdx.x;
  int per = Ksrc * N;
  if (t >= layers * per) return;
  int l = t / per, r = t % per;
  int k = r / N, n = r % N;
  dst[(size_t)l * N * Kdst + (size_t)n * Kdst + k] = f2bf(src[t]);
}

__global__ void k_edge_dist(const float* __restrict__ pos, const int* __restrict__ ei,
                            float* __restrict__ dist) {
  int e = blockIdx.x * blockDim.x + threadIdx.x;
  if (e >= NEDGE) return;
  int s = ei[e], d = ei[NEDGE + e];
  float dx = pos[d * 3 + 0] - pos[s * 3 + 0];
  float dy = pos[d * 3 + 1] - pos[s * 3 + 1];
  float dz = pos[d * 3 + 2] - pos[s * 3 + 2];
  dist[e] = sqrtf(dx * dx + dy * dy + dz * dz);
}

// RBF expansion, padded to 128 columns (cols >= 50 are zero)
__global__ void k_rbf(const float* __restrict__ dist, u16* __restrict__ out) {
  int t = blockIdx.x * blockDim.x + threadIdx.x;
  if (t >= NEDGE * 128) return;
  int e = t >> 7, c = t & 127;
  float val = 0.0f;
  if (c < 50) {
    float d = dist[e];
    float center = (8.0f / 49.0f) * (float)c;
    float w = 8.0f / 50.0f;
    float x = d - center;
    float r = __expf(-(x * x) / (2.0f * w * w));
    float cut = (d < 8.0f) ? 0.5f * (__cosf(d * FPI / 8.0f) + 1.0f) : 0.0f;
    val = r * cut;
  }
  out[t] = f2bf(val);
}

__global__ void k_line_init(const float* __restrict__ ang, const float* __restrict__ w,
                            const float* __restrict__ b, float* lf, u16* lb) {
  int t = blockIdx.x * blockDim.x + threadIdx.x;
  if (t >= NLINE * 128) return;
  int row = t >> 7, c = t & 127;
  float v = ang[row] * w[c] + b[c];
  lf[t] = v;
  lb[t] = f2bf(v);
}

__global__ void k_node_init(const int* __restrict__ z, const float* __restrict__ tab,
                            float* nf, u16* nb) {
  int t = blockIdx.x * blockDim.x + threadIdx.x;
  if (t >= NNODE * 128) return;
  int row = t >> 7, c = t & 127;
  float v = tab[(size_t)(z[row] - 1) * 128 + c];
  nf[t] = v;
  nb[t] = f2bf(v);
}

// scatter-add for segment mean (src rows -> accum[idx[row]])
__global__ void k_scatter_add(const float* __restrict__ src, const int* __restrict__ idx,
                              float* accum, float* cnt, int nrows) {
  int t = blockIdx.x * blockDim.x + threadIdx.x;
  if (t >= nrows * 128) return;
  int row = t >> 7, c = t & 127;
  int d = idx[row];
  atomicAdd(&accum[(size_t)d * 128 + c], src[(size_t)row * 128 + c]);
  if (c == 0) atomicAdd(&cnt[d], 1.0f);
}

__global__ void k_seg_div(const float* __restrict__ accum, const float* __restrict__ cnt,
                          u16* outB, int nseg) {
  int t = blockIdx.x * blockDim.x + threadIdx.x;
  if (t >= nseg * 128) return;
  int row = t >> 7;
  float cc = cnt[row];
  if (cc < 1.0f) cc = 1.0f;
  outB[t] = f2bf(accum[t] / cc);
}

// residual + LayerNorm over D=128, one wave per row, 4 elems/lane
__global__ void __launch_bounds__(256) k_res_ln(float* __restrict__ x, const float* __restrict__ upd,
                                                const float* __restrict__ g, const float* __restrict__ b,
                                                u16* __restrict__ xb) {
  int row = blockIdx.x * 8 + (threadIdx.x >> 5);
  int lane = threadIdx.x & 31;
  size_t base = (size_t)row * 128;
  float v[4];
  float m = 0.0f;
#pragma unroll
  for (int j = 0; j < 4; ++j) {
    int c = lane + j * 32;
    v[j] = x[base + c] + upd[base + c];
    m += v[j];
  }
#pragma unroll
  for (int o = 16; o; o >>= 1) m += __shfl_xor(m, o, 32);
  m *= (1.0f / 128.0f);
  float var = 0.0f;
#pragma unroll
  for (int j = 0; j < 4; ++j) { float d = v[j] - m; var += d * d; }
#pragma unroll
  for (int o = 16; o; o >>= 1) var += __shfl_xor(var, o, 32);
  float rs = rsqrtf(var * (1.0f / 128.0f) + 1e-5f);
#pragma unroll
  for (int j = 0; j < 4; ++j) {
    int c = lane + j * 32;
    float y = (v[j] - m) * rs * g[c] + b[c];
    x[base + c] = y;
    xb[base + c] = f2bf(y);
  }
}

// ---------------- generic gather-concat GEMM with WMMA ----------------
// out[M, Nfull] = act( concat_seg(A) @ W + bias ),  W given TRANSPOSED: Wt[Nfull][K] bf16.
// Segment s supplies 128 K-columns: A_row = srcs[s] + (idx?idx[row]:row)*st_s.
// Block: 256 threads = 8 waves; wave -> 16 rows; block covers 128 rows x 128 cols.
// Weight k-slabs (already [n][k] layout) are staged into LDS with the gfx1250
// async global->LDS copy path (ASYNCcnt), then consumed as contiguous B frags.
template <int NSEG>
__global__ void __launch_bounds__(256) k_gemm(
    const u16* __restrict__ s0, const int* __restrict__ i0, int st0,
    const u16* __restrict__ s1, const int* __restrict__ i1, int st1,
    const u16* __restrict__ s2, const int* __restrict__ i2, int st2,
    const u16* __restrict__ Wt, const float* __restrict__ bias,
    float* outF, u16* outB, int M, int Nfull, int act) {
  __shared__ u16 ldsW[128 * 32];  // [n][k] slab, 8 KB
  constexpr int K = NSEG * 128;
  const int tid = threadIdx.x;
  const int lane = tid & 31;
  const int w = tid >> 5;
  const int col = lane & 15, half = lane >> 4;
  const int rowBase = blockIdx.x * 128 + w * 16;
  const int nb = blockIdx.y * 128;
  const int row = rowBase + col;  // A-matrix row handled by this lane (both K halves)

  size_t aoff[3];
  aoff[0] = (size_t)(i0 ? i0[row] : row) * (size_t)st0;
  if (NSEG > 1) aoff[1] = (size_t)(i1 ? i1[row] : row) * (size_t)st1;
  if (NSEG > 2) aoff[2] = (size_t)(i2 ? i2[row] : row) * (size_t)st2;

  v8f acc[8];
#pragma unroll
  for (int i = 0; i < 8; ++i) acc[i] = vzero();

  // async staging assignment: 512 chunks of 16B per 8KB slab, 2 per thread
  const int cn0 = tid >> 2, ck0 = (tid & 3) * 8;                  // chunk 0: n, k-offset (u16)
  const int cn1 = (tid + 256) >> 2, ck1 = ((tid + 256) & 3) * 8;  // chunk 1

#pragma unroll
  for (int s = 0; s < NSEG; ++s) {
    const u16* abase = (s == 0) ? s0 : ((s == 1) ? s1 : s2);
    const u16* arow = abase + aoff[s];
    for (int kk = 0; kk < 128; kk += 32) {
      const int kb = s * 128 + kk;
      __syncthreads();
      {  // async global->LDS copy of Wt[nb..nb+128][kb..kb+32] (row-contiguous)
        unsigned ld0 = (unsigned)(uintptr_t)&ldsW[cn0 * 32 + ck0];
        unsigned long long ga0 =
            (unsigned long long)(uintptr_t)(Wt + (size_t)(nb + cn0) * K + kb + ck0);
        asm volatile("global_load_async_to_lds_b128 %0, %1, off" ::"v"(ld0), "v"(ga0)
                     : "memory");
        unsigned ld1 = (unsigned)(uintptr_t)&ldsW[cn1 * 32 + ck1];
        unsigned long long ga1 =
            (unsigned long long)(uintptr_t)(Wt + (size_t)(nb + cn1) * K + kb + ck1);
        asm volatile("global_load_async_to_lds_b128 %0, %1, off" ::"v"(ld1), "v"(ga1)
                     : "memory");
        asm volatile("s_wait_asynccnt 0" ::: "memory");
      }
      __syncthreads();
      v16bf a = ldv16(arow + kk + half * 16);
#pragma unroll
      for (int nt = 0; nt < 8; ++nt) {
        v16bf bfrag = ldv16(&ldsW[(nt * 16 + col) * 32 + half * 16]);
        acc[nt] = __builtin_amdgcn_wmma_f32_16x16x32_bf16(
            false, a, false, bfrag, (short)0, acc[nt], false, false);
      }
    }
  }

  const int rEpi = rowBase + half * 8;  // C layout: lanes>=16 hold M=8..15
#pragma unroll
  for (int nt = 0; nt < 8; ++nt) {
    const int c = nb + nt * 16 + col;
    const float bb = bias ? bias[c] : 0.0f;
#pragma unroll
    for (int r = 0; r < 8; ++r) {
      float v = acc[nt][r] + bb;
      if (act == 1) v = siluf(v);
      const size_t o = (size_t)(rEpi + r) * (size_t)Nfull + c;
      if (outF) outF[o] = v;
      if (outB) outB[o] = f2bf(v);
    }
  }
}

// ---------------- attention ----------------
// pack qkv f32 [N,384] -> q [H][N][32] bf16, k [H][N][32] bf16, vT [H][32][N] bf16
__global__ void k_qkv_pack(const float* __restrict__ qkv, u16* qp, u16* kp, u16* vT) {
  int t = blockIdx.x * blockDim.x + threadIdx.x;
  if (t >= NNODE * 384) return;
  int n = t / 384, c = t % 384;
  u16 v = f2bf(qkv[t]);
  if (c < 128) {
    int h = c >> 5, d = c & 31;
    qp[((size_t)h * NNODE + n) * 32 + d] = v;
  } else if (c < 256) {
    int cc = c - 128, h = cc >> 5, d = cc & 31;
    kp[((size_t)h * NNODE + n) * 32 + d] = v;
  } else {
    int cc = c - 256, h = cc >> 5, d = cc & 31;
    vT[((size_t)h * 32 + d) * NNODE + n] = v;
  }
}

// S[h][q][k] = (Q K^T)/sqrt(32); grid (kblk=16, qsuper=16, H), 8 waves/block
__global__ void __launch_bounds__(256) k_attn_scores(const u16* __restrict__ qp,
                                                     const u16* __restrict__ kp,
                                                     float* __restrict__ S) {
  const int lane = threadIdx.x & 31, w = threadIdx.x >> 5;
  const int col = lane & 15, half = lane >> 4;
  const int h = blockIdx.z;
  const int q0 = blockIdx.y * 128 + w * 16;
  const int k0 = blockIdx.x * 128;
  const u16* qb = qp + (size_t)h * NNODE * 32;
  const u16* kb = kp + (size_t)h * NNODE * 32;
  v16bf a = ldv16(qb + (size_t)(q0 + col) * 32 + half * 16);
  v8f acc[8];
#pragma unroll
  for (int i = 0; i < 8; ++i) acc[i] = vzero();
#pragma unroll
  for (int nt = 0; nt < 8; ++nt) {
    v16bf bfrag = ldv16(kb + (size_t)(k0 + nt * 16 + col) * 32 + half * 16);
    acc[nt] = __builtin_amdgcn_wmma_f32_16x16x32_bf16(
        false, a, false, bfrag, (short)0, acc[nt], false, false);
  }
  const float scale = 0.17677669529663689f;  // 1/sqrt(32)
  float* Sr = S + (size_t)h * NNODE * NNODE;
  const int rEpi = q0 + half * 8;
#pragma unroll
  for (int nt = 0; nt < 8; ++nt)
#pragma unroll
    for (int r = 0; r < 8; ++r)
      Sr[(size_t)(rEpi + r) * NNODE + k0 + nt * 16 + col] = acc[nt][r] * scale;
}

// row softmax over 2048 cols, one wave per row; writes bf16 attn
__global__ void __launch_bounds__(256) k_softmax(const float* __restrict__ S, u16* __restrict__ A) {
  int row = blockIdx.x * 8 + (threadIdx.x >> 5);
  int lane = threadIdx.x & 31;
  const float* sr = S + (size_t)row * NNODE;
  float mx = -1e30f;
  for (int c = lane; c < NNODE; c += 32) mx = fmaxf(mx, sr[c]);
#pragma unroll
  for (int o = 16; o; o >>= 1) mx = fmaxf(mx, __shfl_xor(mx, o, 32));
  float sum = 0.0f;
  for (int c = lane; c < NNODE; c += 32) sum += __expf(sr[c] - mx);
#pragma unroll
  for (int o = 16; o; o >>= 1) sum += __shfl_xor(sum, o, 32);
  float inv = 1.0f / sum;
  u16* ar = A + (size_t)row * NNODE;
  for (int c = lane; c < NNODE; c += 32) ar[c] = f2bf(__expf(sr[c] - mx) * inv);
}

// O = attn @ V ; grid (qsuper=16, 1, H); writes o bf16 [N][128] at [:, h*32..]
__global__ void __launch_bounds__(256) k_attn_out(const u16* __restrict__ attn,
                                                  const u16* __restrict__ vT,
                                                  u16* __restrict__ oB) {
  const int lane = threadIdx.x & 31, w = threadIdx.x >> 5;
  const int col = lane & 15, half = lane >> 4;
  const int h = blockIdx.z;
  const int q0 = blockIdx.x * 128 + w * 16;
  const u16* ab = attn + (size_t)h * NNODE * NNODE;
  const u16* vb = vT + (size_t)h * 32 * NNODE;
  v8f acc0 = vzero(), acc1 = vzero();
  for (int kb = 0; kb < NNODE; kb += 32) {
    v16bf a = ldv16(ab + (size_t)(q0 + col) * NNODE + kb + half * 16);
    v16bf b0 = ldv16(vb + (size_t)col * NNODE + kb + half * 16);
    v16bf b1 = ldv16(vb + (size_t)(16 + col) * NNODE + kb + half * 16);
    acc0 = __builtin_amdgcn_wmma_f32_16x16x32_bf16(false, a, false, b0, (short)0, acc0, false, false);
    acc1 = __builtin_amdgcn_wmma_f32_16x16x32_bf16(false, a, false, b1, (short)0, acc1, false, false);
  }
  const int rEpi = q0 + half * 8;
#pragma unroll
  for (int r = 0; r < 8; ++r) {
    oB[(size_t)(rEpi + r) * 128 + h * 32 + col]      = f2bf(acc0[r]);
    oB[(size_t)(rEpi + r) * 128 + h * 32 + 16 + col] = f2bf(acc1[r]);
  }
}

// ---------------- readout (tiny MLP, single block) ----------------
__global__ void __launch_bounds__(128) k_readout(const float* __restrict__ node,
                                                 const float* w1, const float* b1,
                                                 const float* w2, const float* b2,
                                                 const float* w3, const float* b3,
                                                 float* out) {
  __shared__ float g[128], h1[128], h2[64];
  int t = threadIdx.x;
  float s = 0.0f;
  for (int n = 0; n < NNODE; ++n) s += node[(size_t)n * 128 + t];
  g[t] = s * (1.0f / (float)NNODE);
  __syncthreads();
  float a = b1[t];
  for (int k = 0; k < 128; ++k) a += g[k] * w1[k * 128 + t];
  h1[t] = siluf(a);
  __syncthreads();
  if (t < 64) {
    float a2 = b2[t];
    for (int k = 0; k < 128; ++k) a2 += h1[k] * w2[k * 64 + t];
    h2[t] = siluf(a2);
  }
  __syncthreads();
  if (t == 0) {
    float a3 = b3[0];
    for (int k = 0; k < 64; ++k) a3 += h2[k] * w3[k];
    out[0] = a3;
  }
}

// ---------------- host launch ----------------
extern "C" void kernel_launch(void* const* d_in, const int* in_sizes, int n_in,
                              void* d_out, int out_size, void* d_ws, size_t ws_size,
                              hipStream_t stream) {
  (void)in_sizes; (void)n_in; (void)out_size; (void)ws_size;

  const int*   atomic_numbers = (const int*)d_in[0];
  const float* positions      = (const float*)d_in[1];
  const int*   edge_index     = (const int*)d_in[2];
  const int*   line_edge_idx  = (const int*)d_in[3];
  const float* line_angles    = (const float*)d_in[4];
  const float* atom_table     = (const float*)d_in[5];
  const float* rbf_w1 = (const float*)d_in[6];
  const float* rbf_b1 = (const float*)d_in[7];
  const float* rbf_w2 = (const float*)d_in[8];
  const float* rbf_b2 = (const float*)d_in[9];
  const float* line_emb_w = (const float*)d_in[10];
  const float* line_emb_b = (const float*)d_in[11];
  const float* eu_w1 = (const float*)d_in[12];
  const float* eu_b1 = (const float*)d_in[13];
  const float* eu_w2 = (const float*)d_in[14];
  const float* eu_b2 = (const float*)d_in[15];
  const float* lu_w1 = (const float*)d_in[16];
  const float* lu_b1 = (const float*)d_in[17];
  const float* lu_w2 = (const float*)d_in[18];
  const float* lu_b2 = (const float*)d_in[19];
  const float* nu_w1 = (const float*)d_in[20];
  const float* nu_b1 = (const float*)d_in[21];
  const float* nu_w2 = (const float*)d_in[22];
  const float* nu_b2 = (const float*)d_in[23];
  const float* attn_wqkv = (const float*)d_in[24];
  const float* attn_bqkv = (const float*)d_in[25];
  const float* attn_wo = (const float*)d_in[26];
  const float* attn_bo = (const float*)d_in[27];
  const float* nn_g = (const float*)d_in[28];
  const float* nn_b = (const float*)d_in[29];
  const float* en_g = (const float*)d_in[30];
  const float* en_b = (const float*)d_in[31];
  const float* ln_g = (const float*)d_in[32];
  const float* ln_b = (const float*)d_in[33];
  const float* ro_w1 = (const float*)d_in[34];
  const float* ro_b1 = (const float*)d_in[35];
  const float* ro_w2 = (const float*)d_in[36];
  const float* ro_b2 = (const float*)d_in[37];
  const float* ro_w3 = (const float*)d_in[38];
  const float* ro_b3 = (const float*)d_in[39];

  const int* edst = edge_index + NEDGE;
  const int* lsrc = line_edge_idx;
  const int* ldst = line_edge_idx + NLINE;

  // ---- workspace carving ----
  char* base = (char*)d_ws;
  size_t off = 0;
  auto alloc = [&](size_t bytes) -> void* {
    void* p = base + off;
    off += (bytes + 255) & ~(size_t)255;
    return p;
  };
  // all weight blocks stored TRANSPOSED: [layers][N][K] bf16
  u16* rbfw1t = (u16*)alloc(128 * 128 * 2);  // padded K: [128n][128k]
  u16* rbfw2t = (u16*)alloc(128 * 128 * 2);
  u16* luw1t  = (u16*)alloc((size_t)NLAY * 256 * 128 * 2);
  u16* luw2t  = (u16*)alloc((size_t)NLAY * 128 * 128 * 2);
  u16* euw1t  = (u16*)alloc((size_t)NLAY * 384 * 256 * 2);
  u16* euw2t  = (u16*)alloc((size_t)NLAY * 256 * 128 * 2);
  u16* nuw1t  = (u16*)alloc((size_t)NLAY * 256 * 256 * 2);
  u16* nuw2t  = (u16*)alloc((size_t)NLAY * 256 * 128 * 2);
  u16* wqkvt  = (u16*)alloc((size_t)NLAY * 128 * 384 * 2);
  u16* wot    = (u16*)alloc((size_t)NLAY * 128 * 128 * 2);

  float* node_f = (float*)alloc((size_t)NNODE * 128 * 4);
  u16*   node_b = (u16*)alloc((size_t)NNODE * 128 * 2);
  float* edge_f = (float*)alloc((size_t)NEDGE * 128 * 4);
  u16*   edge_b = (u16*)alloc((size_t)NEDGE * 128 * 2);
  float* line_f = (float*)alloc((size_t)NLINE * 128 * 4);
  u16*   line_b = (u16*)alloc((size_t)NLINE * 128 * 2);
  float* dist   = (float*)alloc((size_t)NEDGE * 4);
  u16*   rbfpad = (u16*)alloc((size_t)NEDGE * 128 * 2);
  u16*   hscr   = (u16*)alloc((size_t)NLINE * 128 * 2);   // hidden scratch (biggest user: lu hidden)
  float* tmp    = (float*)alloc((size_t)NLINE * 128 * 4); // pre-LN update scratch / qkv f32
  float* accum  = (float*)alloc((size_t)NEDGE * 128 * 4);
  float* cnt    = (float*)alloc((size_t)NEDGE * 4);
  u16*   aggr_b = (u16*)alloc((size_t)NEDGE * 128 * 2);
  u16*   qp     = (u16*)alloc((size_t)NH * NNODE * 32 * 2);
  u16*   kp     = (u16*)alloc((size_t)NH * NNODE * 32 * 2);
  u16*   vT     = (u16*)alloc((size_t)NH * NNODE * 32 * 2);
  float* scores = (float*)alloc((size_t)NH * NNODE * NNODE * 4);
  u16*   attnB  = (u16*)alloc((size_t)NH * NNODE * NNODE * 2);
  u16*   oB     = (u16*)alloc((size_t)NNODE * 128 * 2);

  auto cvtT = [&](u16* dst, const float* src, int Ksrc, int N, int Kdst, int layers) {
    int tot = layers * Ksrc * N;
    k_f2bt<<<(tot + 255) / 256, 256, 0, stream>>>(dst, src, Ksrc, N, Kdst, layers);
  };
  auto gemm = [&](const u16* s0, const int* i0, int st0,
                  const u16* s1, const int* i1, int st1,
                  const u16* s2, const int* i2, int st2,
                  int nseg, const u16* Wt, const float* bias,
                  float* oF, u16* oBf, int M, int Nf, int act) {
    dim3 grid(M / 128, Nf / 128, 1);
    if (nseg == 1)
      k_gemm<1><<<grid, 256, 0, stream>>>(s0, i0, st0, s1, i1, st1, s2, i2, st2,
                                          Wt, bias, oF, oBf, M, Nf, act);
    else if (nseg == 2)
      k_gemm<2><<<grid, 256, 0, stream>>>(s0, i0, st0, s1, i1, st1, s2, i2, st2,
                                          Wt, bias, oF, oBf, M, Nf, act);
    else
      k_gemm<3><<<grid, 256, 0, stream>>>(s0, i0, st0, s1, i1, st1, s2, i2, st2,
                                          Wt, bias, oF, oBf, M, Nf, act);
  };

  // ---- convert weights to bf16, transposed [N][K] (deterministic each call) ----
  hipMemsetAsync(rbfw1t, 0, 128 * 128 * 2, stream);
  cvtT(rbfw1t, rbf_w1, 50, 128, 128, 1);  // K padded 50 -> 128
  cvtT(rbfw2t, rbf_w2, 128, 128, 128, 1);
  cvtT(luw1t, lu_w1, 256, 128, 256, NLAY);
  cvtT(luw2t, lu_w2, 128, 128, 128, NLAY);
  cvtT(euw1t, eu_w1, 384, 256, 384, NLAY);
  cvtT(euw2t, eu_w2, 256, 128, 256, NLAY);
  cvtT(nuw1t, nu_w1, 256, 256, 256, NLAY);
  cvtT(nuw2t, nu_w2, 256, 128, 256, NLAY);
  cvtT(wqkvt, attn_wqkv, 128, 384, 128, NLAY);
  cvtT(wot, attn_wo, 128, 128, 128, NLAY);

  // ---- embeddings ----
  k_edge_dist<<<NEDGE / 256, 256, 0, stream>>>(positions, edge_index, dist);
  k_rbf<<<(NEDGE * 128) / 256, 256, 0, stream>>>(dist, rbfpad);
  gemm(rbfpad, nullptr, 128, nullptr, nullptr, 0, nullptr, nullptr, 0, 1,
       rbfw1t, rbf_b1, nullptr, hscr, NEDGE, 128, 1);
  gemm(hscr, nullptr, 128, nullptr, nullptr, 0, nullptr, nullptr, 0, 1,
       rbfw2t, rbf_b2, edge_f, edge_b, NEDGE, 128, 0);
  k_line_init<<<(NLINE * 128) / 256, 256, 0, stream>>>(line_angles, line_emb_w, line_emb_b, line_f, line_b);
  k_node_init<<<(NNODE * 128) / 256, 256, 0, stream>>>(atomic_numbers, atom_table, node_f, node_b);

  // ---- layers ----
  for (int l = 0; l < NLAY; ++l) {
    const u16* luw1 = luw1t + (size_t)l * 256 * 128;
    const u16* luw2 = luw2t + (size_t)l * 128 * 128;
    const u16* euw1 = euw1t + (size_t)l * 384 * 256;
    const u16* euw2 = euw2t + (size_t)l * 256 * 128;
    const u16* nuw1 = nuw1t + (size_t)l * 256 * 256;
    const u16* nuw2 = nuw2t + (size_t)l * 256 * 128;
    const u16* wq   = wqkvt + (size_t)l * 128 * 384;
    const u16* wo   = wot + (size_t)l * 128 * 128;

    // line update: lu = silu([e_src, e_dst] @ W1 + b1) @ W2 + b2 ; line = LN(line + lu)
    gemm(edge_b, lsrc, 128, edge_b, ldst, 128, nullptr, nullptr, 0, 2,
         luw1, lu_b1 + l * 128, nullptr, hscr, NLINE, 128, 1);
    gemm(hscr, nullptr, 128, nullptr, nullptr, 0, nullptr, nullptr, 0, 1,
         luw2, lu_b2 + l * 128, tmp, nullptr, NLINE, 128, 0);
    k_res_ln<<<NLINE / 8, 256, 0, stream>>>(line_f, tmp, ln_g + l * 128, ln_b + l * 128, line_b);

    // segment mean line -> edges
    hipMemsetAsync(accum, 0, (size_t)NEDGE * 128 * 4, stream);
    hipMemsetAsync(cnt, 0, (size_t)NEDGE * 4, stream);
    k_scatter_add<<<(NLINE * 128) / 256, 256, 0, stream>>>(line_f, ldst, accum, cnt, NLINE);
    k_seg_div<<<(NEDGE * 128) / 256, 256, 0, stream>>>(accum, cnt, aggr_b, NEDGE);

    // edge update: eu = silu([e, e, laggr] @ W1 + b1) @ W2 + b2 ; edge = LN(edge + eu)
    gemm(edge_b, nullptr, 128, edge_b, nullptr, 128, aggr_b, nullptr, 128, 3,
         euw1, eu_b1 + l * 256, nullptr, hscr, NEDGE, 256, 1);
    gemm(hscr, nullptr, 256, hscr + 128, nullptr, 256, nullptr, nullptr, 0, 2,
         euw2, eu_b2 + l * 128, tmp, nullptr, NEDGE, 128, 0);
    k_res_ln<<<NEDGE / 8, 256, 0, stream>>>(edge_f, tmp, en_g + l * 128, en_b + l * 128, edge_b);

    // segment mean edges -> nodes
    hipMemsetAsync(accum, 0, (size_t)NNODE * 128 * 4, stream);
    hipMemsetAsync(cnt, 0, (size_t)NNODE * 4, stream);
    k_scatter_add<<<(NEDGE * 128) / 256, 256, 0, stream>>>(edge_f, edst, accum, cnt, NEDGE);
    k_seg_div<<<(NNODE * 128) / 256, 256, 0, stream>>>(accum, cnt, aggr_b, NNODE);

    // node update: nu = silu([n, naggr] @ W1 + b1) @ W2 + b2 ; node = LN(node + nu)
    gemm(node_b, nullptr, 128, aggr_b, nullptr, 128, nullptr, nullptr, 0, 2,
         nuw1, nu_b1 + l * 256, nullptr, hscr, NNODE, 256, 1);
    gemm(hscr, nullptr, 256, hscr + 128, nullptr, 256, nullptr, nullptr, 0, 2,
         nuw2, nu_b2 + l * 128, tmp, nullptr, NNODE, 128, 0);
    k_res_ln<<<NNODE / 8, 256, 0, stream>>>(node_f, tmp, nn_g + l * 128, nn_b + l * 128, node_b);

    // global self-attention: node = (softmax(QK^T/sqrt(dh)) V) @ Wo + bo
    gemm(node_b, nullptr, 128, nullptr, nullptr, 0, nullptr, nullptr, 0, 1,
         wq, attn_bqkv + (size_t)l * 384, tmp, nullptr, NNODE, 384, 0);
    k_qkv_pack<<<(NNODE * 384) / 256, 256, 0, stream>>>(tmp, qp, kp, vT);
    k_attn_scores<<<dim3(16, 16, NH), 256, 0, stream>>>(qp, kp, scores);
    k_softmax<<<(NH * NNODE) / 8, 256, 0, stream>>>(scores, attnB);
    k_attn_out<<<dim3(16, 1, NH), 256, 0, stream>>>(attnB, vT, oB);
    gemm(oB, nullptr, 128, nullptr, nullptr, 0, nullptr, nullptr, 0, 1,
         wo, attn_bo + l * 128, node_f, node_b, NNODE, 128, 0);
  }

  // ---- readout ----
  k_readout<<<1, 128, 0, stream>>>(node_f, ro_w1, ro_b1, ro_w2, ro_b2, ro_w3, ro_b3,
                                   (float*)d_out);
}
